// Model_62758062129476
// MI455X (gfx1250) — matrix operand order repaired and checked by reading, not compiled, not run on hardware
//
#include <hip/hip_runtime.h>
#include <math.h>

// ---------------------------------------------------------------------------
// CDNA5 (gfx1250, wave32) WMMA types & fragment helpers
// Layouts used here (A/B k order is this file's contiguous one, not the ISA table's):
//   A (16x32 f16): M = lane%16; VGPR r holds K pair:
//       2r (+16 if lane>=16)
//   B (32x16 f16): N = lane%16; VGPR r holds K = 2r,2r+1 (+16 if lane>=16)
//   C/D (16x16 f32): VGPR r -> M = r + 8*(lane>=16), N = lane%16
// All fragment loads below read CONTIGUOUS 16-half rows -> ds_load_b128 pairs.
// ---------------------------------------------------------------------------
typedef __attribute__((ext_vector_type(16))) _Float16 v16h;
typedef __attribute__((ext_vector_type(8)))  float    v8f;

#define WMMA(a, b, c) \
  __builtin_amdgcn_wmma_f32_16x16x32_f16(false, (a), false, (b), (short)0, (c), false, false)

__device__ __forceinline__ int ka_pair(int r, int lane) {
  int kp = 2 * r;
  return kp + ((lane >> 4) ? 16 : 0);
}

// A[m][k] = p[(m0+m)*stride + k0+k]
__device__ __forceinline__ v16h frag_a(const _Float16* p, int stride, int m0, int k0) {
  int lane = threadIdx.x & 31;
  const _Float16* row = p + (size_t)(m0 + (lane & 15)) * stride + k0;
  v16h a;
#pragma unroll
  for (int r = 0; r < 8; ++r) {
    int kp = ka_pair(r, lane);
    a[2 * r]     = row[kp];
    a[2 * r + 1] = row[kp + 1];
  }
  return a;
}

// A[m][k] = p[(m0+m)*stride + k0+k] * s   (s constant per lane's row)
__device__ __forceinline__ v16h frag_a_scaled(const _Float16* p, int stride, int m0, int k0, float s) {
  int lane = threadIdx.x & 31;
  const _Float16* row = p + (size_t)(m0 + (lane & 15)) * stride + k0;
  v16h a;
#pragma unroll
  for (int r = 0; r < 8; ++r) {
    int kp = ka_pair(r, lane);
    a[2 * r]     = (_Float16)((float)row[kp] * s);
    a[2 * r + 1] = (_Float16)((float)row[kp + 1] * s);
  }
  return a;
}

// B[k][n] = p[(n0+n)*stride + k0+k]   (memory holds B^T row-major)
__device__ __forceinline__ v16h frag_b_t(const _Float16* p, int stride, int k0, int n0) {
  int lane = threadIdx.x & 31;
  const _Float16* row = p + (size_t)(n0 + (lane & 15)) * stride + k0 + ((lane >> 4) ? 16 : 0);
  v16h b;
#pragma unroll
  for (int r = 0; r < 8; ++r) {
    b[2 * r]     = row[2 * r];
    b[2 * r + 1] = row[2 * r + 1];
  }
  return b;
}

__device__ __forceinline__ float leaky(float v) { return v > 0.f ? v : 0.01f * v; }

// ---------------------------------------------------------------------------
// LayerNorm kernels (wave32: one warp per 64-row, half-warp per 16-row)
// ---------------------------------------------------------------------------
__global__ __launch_bounds__(256) void ln64_kernel(const float* __restrict__ in0,
                                                   const float* __restrict__ in1,
                                                   const float* __restrict__ g,
                                                   const float* __restrict__ b,
                                                   float* __restrict__ out, int rows) {
  int warp = blockIdx.x * (blockDim.x >> 5) + (threadIdx.x >> 5);
  if (warp >= rows) return;
  int lane = threadIdx.x & 31;
  const float* r0 = in0 + (size_t)warp * 64;
  float x0 = r0[lane], x1 = r0[lane + 32];
  if (in1) {
    const float* r1 = in1 + (size_t)warp * 64;
    x0 += r1[lane];
    x1 += r1[lane + 32];
  }
  float s = x0 + x1, sq = x0 * x0 + x1 * x1;
#pragma unroll
  for (int m = 16; m > 0; m >>= 1) {
    s  += __shfl_xor(s, m, 32);
    sq += __shfl_xor(sq, m, 32);
  }
  float mean = s * (1.f / 64.f);
  float var = sq * (1.f / 64.f) - mean * mean;
  float inv = rsqrtf(var + 1e-5f);
  float* o = out + (size_t)warp * 64;
  o[lane]      = (x0 - mean) * inv * g[lane] + b[lane];
  o[lane + 32] = (x1 - mean) * inv * g[lane + 32] + b[lane + 32];
}

__global__ __launch_bounds__(256) void ln16_kernel(const float* __restrict__ in0,
                                                   const float* __restrict__ in1,
                                                   const float* __restrict__ g,
                                                   const float* __restrict__ b,
                                                   float* __restrict__ out, int rows) {
  int row = blockIdx.x * (blockDim.x >> 4) + (threadIdx.x >> 4);
  if (row >= rows) return;
  int lane = threadIdx.x & 15;
  float x = in0[(size_t)row * 16 + lane];
  if (in1) x += in1[(size_t)row * 16 + lane];
  float s = x, sq = x * x;
#pragma unroll
  for (int m = 8; m > 0; m >>= 1) {
    s  += __shfl_xor(s, m, 32);
    sq += __shfl_xor(sq, m, 32);
  }
  float mean = s * (1.f / 16.f);
  float var = sq * (1.f / 16.f) - mean * mean;
  float inv = rsqrtf(var + 1e-5f);
  out[(size_t)row * 16 + lane] = (x - mean) * inv * g[lane] + b[lane];
}

// ---------------------------------------------------------------------------
// Lightning attention (q=k=v=x), chunked scan: one workgroup per (b,h),
// 8 wave32 waves, state S kept f32 in LDS across 20 chunks of 150.
// Per chunk, transposed f16 copies are staged once so every WMMA fragment
// load is contiguous (ds_load_b128):
//   xs  [CT][ST]  : x[t][d]              (A of sc, A of o-term2, B^T of sc)
//   xsT [D][CTS]  : x[u][d] transposed   (B^T of o-term1, A of S-update)
//   vdT [D][CTS]  : x[u][e]*kdec[u] tr.  (B^T of S-update; kdec chunk-invariant)
//   ShT [D][SKT]  : f16 shadow of S^T    (B^T of o-term2)
// Decay mask via full (negative-lag-padded) table -> branchless.
// ---------------------------------------------------------------------------
struct SlopeArr { float s[16]; };

template <int D>
__global__ __launch_bounds__(256) void attn_kernel(const float* __restrict__ x,
                                                   float* __restrict__ out,
                                                   SlopeArr slopes, int H) {
  constexpr int C   = 150;                 // chunk
  constexpr int CT  = 160;                 // padded chunk (10 tiles)
  constexpr int KP  = (D < 32) ? 32 : D;   // padded feature K
  constexpr int ST  = KP + 8;              // xs stride
  constexpr int SC  = CT + 8;              // score stride
  constexpr int CTS = CT + 8;              // transposed-buffer stride
  constexpr int SKT = KP + 8;              // S-shadow stride
  constexpr int ND  = D / 16;              // feature tiles

  extern __shared__ char smemraw[];
  _Float16* xs   = (_Float16*)smemraw;        // [CT][ST]
  _Float16* scs  = xs + CT * ST;              // [CT][SC]
  _Float16* xsT  = scs + CT * SC;             // [D][CTS]
  _Float16* vdT  = xsT + D * CTS;             // [D][CTS]
  _Float16* ShT  = vdT + D * CTS;             // [D][SKT]
  float*    Sf   = (float*)(ShT + D * SKT);   // [D][D]
  float*    decf = Sf + D * D;                // [2*CT]  lag table, 0 for lag<0
  float*    qdec = decf + 2 * CT;             // [CT]
  float*    kdec = qdec + CT;                 // [CT]

  int bh = blockIdx.x;
  int h = bh % H;
  float logl = -slopes.s[h];
  float blk = __expf(logl * (float)C);
  int tid = threadIdx.x, wave = tid >> 5, lane = tid & 31;

  for (int i = tid; i < 2 * CT; i += 256)
    decf[i] = (i >= CT - 1) ? __expf(logl * (float)(i - (CT - 1))) : 0.f;
  for (int i = tid; i < CT; i += 256) {
    qdec[i] = __expf(logl * (float)(i + 1));
    kdec[i] = (i < C) ? __expf(logl * (float)(C - 1 - i)) : 0.f;
  }
  for (int i = tid; i < D * D; i += 256) Sf[i] = 0.f;
  for (int i = tid; i < D * SKT; i += 256) ShT[i] = (_Float16)0.f;

  const float* xb = x + (size_t)bh * 3000 * D;
  float* ob = out + (size_t)bh * 3000 * D;
  __syncthreads();

  for (int ch = 0; ch < 20; ++ch) {
    const float* xc = xb + (size_t)ch * C * D;
    // stage chunk as f16 (zero-padded rows C..CT and cols D..KP)
    for (int i = tid; i < CT * ST; i += 256) {
      int r = i / ST, c2 = i % ST;
      float v = (r < C && c2 < D) ? xc[(size_t)r * D + c2] : 0.f;
      xs[i] = (_Float16)v;
    }
    // prefetch next chunk while this one computes (global_prefetch_b8)
    if (ch < 19) __builtin_prefetch(xc + (size_t)C * D + tid * 16, 0, 0);
    __syncthreads();
    // build transposed copies from LDS (contiguous-fragment sources)
    for (int i = tid; i < D * CTS; i += 256) {
      int d = i / CTS, u = i % CTS;
      float v = (u < CT) ? (float)xs[(size_t)u * ST + d] : 0.f;
      xsT[i] = (_Float16)v;
      vdT[i] = (_Float16)(v * ((u < CT) ? kdec[u] : 0.f));
    }
    __syncthreads();

    // sc[t][u] = (q_t . k_u) * decay(t-u)   (100 tiles over 8 waves)
    for (int t = wave; t < 100; t += 8) {
      int mt = (t / 10) * 16, nt = (t % 10) * 16;
      v8f acc = {};
#pragma unroll
      for (int kk = 0; kk < KP; kk += 32) {
        v16h a = frag_a(xs, ST, mt, kk);
        v16h b = frag_b_t(xs, ST, kk, nt);  // B[k=d][n=u] = x[u][d]
        acc = WMMA(a, b, acc);
      }
      int mrow = mt + ((lane >> 4) << 3);
      int n = nt + (lane & 15);
#pragma unroll
      for (int r2 = 0; r2 < 8; ++r2) {
        int m = mrow + r2;
        float w = decf[m - n + (CT - 1)];   // branchless mask
        scs[(size_t)m * SC + n] = (_Float16)(acc[r2] * w);
      }
    }
    __syncthreads();

    // o = sc @ v + (q*qdec) @ S      (10*ND tiles)
    for (int t = wave; t < 10 * ND; t += 8) {
      int mt = (t / ND) * 16, nt = (t % ND) * 16;
      v8f acc = {};
#pragma unroll
      for (int kk = 0; kk < CT; kk += 32) {
        v16h a = frag_a(scs, SC, mt, kk);
        v16h b = frag_b_t(xsT, CTS, kk, nt);  // B[k=u][n=e] = xsT[e][u]
        acc = WMMA(a, b, acc);
      }
      float qs = qdec[mt + (lane & 15)];
#pragma unroll
      for (int kk = 0; kk < KP; kk += 32) {
        v16h a = frag_a_scaled(xs, ST, mt, kk, qs);
        v16h b = frag_b_t(ShT, SKT, kk, nt);  // B[k=d][n=e] = ShT[e][d]
        acc = WMMA(a, b, acc);
      }
      int mrow = mt + ((lane >> 4) << 3);
      int n = nt + (lane & 15);
#pragma unroll
      for (int r2 = 0; r2 < 8; ++r2) {
        int m = mrow + r2;
        if (m < C) ob[(size_t)(ch * C + m) * D + n] = acc[r2];
      }
    }

    // S_acc = k^T @ (kdec * v)       (ND*ND tiles; kept in regs until barrier)
    v8f sacc[2];
    int si = 0;
    for (int t = wave; t < ND * ND; t += 8, ++si) {
      int mt = (t / ND) * 16, nt = (t % ND) * 16;
      v8f acc = {};
#pragma unroll
      for (int kk = 0; kk < CT; kk += 32) {
        v16h a = frag_a(xsT, CTS, mt, kk);    // A[d][u] = xsT[d][u]
        v16h b = frag_b_t(vdT, CTS, kk, nt);  // B[k=u][n=e] = vdT[e][u]
        acc = WMMA(a, b, acc);
      }
      sacc[si] = acc;
    }
    __syncthreads();  // all reads of old S / staged buffers done

    si = 0;
    for (int t = wave; t < ND * ND; t += 8, ++si) {
      int mt = (t / ND) * 16, nt = (t % ND) * 16;
      int mrow = mt + ((lane >> 4) << 3);
      int n = nt + (lane & 15);
#pragma unroll
      for (int r2 = 0; r2 < 8; ++r2) {
        int d = mrow + r2;
        float nv = blk * Sf[(size_t)d * D + n] + sacc[si][r2];
        Sf[(size_t)d * D + n] = nv;
        ShT[(size_t)n * SKT + d] = (_Float16)nv;  // transposed f16 shadow
      }
    }
    __syncthreads();
  }
}

// ---------------------------------------------------------------------------
// Fused FFN: out = leaky(x @ w1^T + b1) @ w2^T + b2, 64-token tiles,
// both weight matrices staged once in LDS as f16.
// ---------------------------------------------------------------------------
template <int DIN, int KPIN, int DHID, int DOUT>
__global__ __launch_bounds__(256) void ff_kernel(const float* __restrict__ xin,
                                                 const float* __restrict__ w1,
                                                 const float* __restrict__ b1,
                                                 const float* __restrict__ w2,
                                                 const float* __restrict__ b2,
                                                 float* __restrict__ outp, long total) {
  constexpr int SW1 = KPIN + 8;
  constexpr int SW2 = DHID + 8;
  constexpr int SX  = KPIN + 8;
  constexpr int SH  = DHID + 8;
  extern __shared__ char smemraw[];
  _Float16* w1s = (_Float16*)smemraw;      // [DHID][SW1]  (w1[h][din])
  _Float16* w2s = w1s + DHID * SW1;        // [DOUT][SW2]  (w2[o][h])
  _Float16* xsh = w2s + DOUT * SW2;        // [64][SX]
  _Float16* hsh = xsh + 64 * SX;           // [64][SH]
  float* b1s = (float*)(hsh + 64 * SH);
  float* b2s = b1s + DHID;

  int tid = threadIdx.x, wave = tid >> 5, lane = tid & 31;
  long row0 = (long)blockIdx.x * 64;

  for (int i = tid; i < DHID * SW1; i += 256) {
    int hh = i / SW1, d = i % SW1;
    w1s[i] = (_Float16)((d < DIN) ? w1[(size_t)hh * DIN + d] : 0.f);
  }
  for (int i = tid; i < DOUT * SW2; i += 256) {
    int o = i / SW2, hh = i % SW2;
    w2s[i] = (_Float16)((hh < DHID) ? w2[(size_t)o * DHID + hh] : 0.f);
  }
  for (int i = tid; i < DHID; i += 256) b1s[i] = b1[i];
  for (int i = tid; i < DOUT; i += 256) b2s[i] = b2[i];
  for (int i = tid; i < 64 * SX; i += 256) {
    int r = i / SX, d = i % SX;
    long gr = row0 + r;
    xsh[i] = (_Float16)((d < DIN && gr < total) ? xin[gr * DIN + d] : 0.f);
  }
  __syncthreads();

  constexpr int NH = DHID / 16;
  for (int t = wave; t < 4 * NH; t += 8) {
    int mt = (t / NH) * 16, nt = (t % NH) * 16;
    v8f acc = {};
#pragma unroll
    for (int kk = 0; kk < KPIN; kk += 32) {
      v16h a = frag_a(xsh, SX, mt, kk);
      v16h b = frag_b_t(w1s, SW1, kk, nt);
      acc = WMMA(a, b, acc);
    }
    int mrow = mt + ((lane >> 4) << 3);
    int n = nt + (lane & 15);
    float bb = b1s[n];
#pragma unroll
    for (int r2 = 0; r2 < 8; ++r2) {
      float v = leaky(acc[r2] + bb);
      hsh[(size_t)(mrow + r2) * SH + n] = (_Float16)v;
    }
  }
  __syncthreads();

  constexpr int NO = DOUT / 16;
  for (int t = wave; t < 4 * NO; t += 8) {
    int mt = (t / NO) * 16, nt = (t % NO) * 16;
    v8f acc = {};
#pragma unroll
    for (int kk = 0; kk < DHID; kk += 32) {
      v16h a = frag_a(hsh, SH, mt, kk);
      v16h b = frag_b_t(w2s, SW2, kk, nt);
      acc = WMMA(a, b, acc);
    }
    int mrow = mt + ((lane >> 4) << 3);
    int n = nt + (lane & 15);
    float bb = b2s[n];
#pragma unroll
    for (int r2 = 0; r2 < 8; ++r2) {
      long gr = row0 + mrow + r2;
      if (gr < total) outp[gr * DOUT + n] = acc[r2] + bb;
    }
  }
}

// ---------------------------------------------------------------------------
// Small glue kernels
// ---------------------------------------------------------------------------
__global__ void preproj_kernel(const float* __restrict__ seqx, const float* __restrict__ w1,
                               const float* __restrict__ b1, const float* __restrict__ w2,
                               const float* __restrict__ b2, const float* __restrict__ pe0,
                               float* __restrict__ out) {
  __shared__ float hid[32];
  __shared__ float xin[8];
  size_t t = blockIdx.x;
  if (threadIdx.x < 8) xin[threadIdx.x] = seqx[t * 8 + threadIdx.x];
  __syncthreads();
  if (threadIdx.x < 32) {
    float a = b1[threadIdx.x];
    for (int k = 0; k < 8; ++k) a += xin[k] * w1[threadIdx.x * 8 + k];
    hid[threadIdx.x] = leaky(a);
  }
  __syncthreads();
  int j = threadIdx.x;
  float a = b2[j];
  for (int k = 0; k < 32; ++k) a += hid[k] * w2[j * 32 + k];
  out[t * 64 + j] = a + pe0[j];
}

__global__ void headmean_kernel(const float* __restrict__ in, float* __restrict__ out,
                                int H, int D, long totalBND) {
  long i = (long)blockIdx.x * blockDim.x + threadIdx.x;
  if (i >= totalBND) return;
  long bn = i / D;
  int d = (int)(i % D);
  long b = bn / 3000, n = bn % 3000;
  float s = 0.f;
  for (int h = 0; h < H; ++h) s += in[((b * H + h) * 3000 + n) * D + d];
  out[i] = s / (float)H;
}

// conv1d(VALID) + leaky, fused reduction to per-(b,c) sum & max
// (SE-scale commutes with max since sigmoid > 0, so full map never stored)
__global__ __launch_bounds__(256) void conv_reduce_kernel(const float* __restrict__ in,
                                                          const float* __restrict__ w,
                                                          const float* __restrict__ bias,
                                                          float* __restrict__ sums,
                                                          float* __restrict__ maxs,
                                                          int C, int L, int ks) {
  __shared__ float wrow[320];
  __shared__ float rs[256], rm[256];
  int bc = blockIdx.x;
  int b = bc / C, c = bc % C;
  int tid = threadIdx.x;
  for (int i = tid; i < C * ks; i += blockDim.x) wrow[i] = w[(size_t)c * C * ks + i];
  __syncthreads();
  int Lout = L - ks + 1;
  float bia = bias[c];
  float bsum = 0.f, bmax = -3.4e38f;
  for (int l = tid; l < Lout; l += blockDim.x) {
    float acc = bia;
    for (int kk = 0; kk < ks; ++kk) {
      const float* ip = in + ((size_t)b * L + l + kk) * C;
      const float* wp = wrow + kk;
      for (int ci = 0; ci < C; ++ci) acc += ip[ci] * wp[ci * ks];
    }
    float h = leaky(acc);
    bsum += h;
    bmax = fmaxf(bmax, h);
  }
  rs[tid] = bsum;
  rm[tid] = bmax;
  __syncthreads();
  for (int s2 = 128; s2 > 0; s2 >>= 1) {
    if (tid < s2) {
      rs[tid] += rs[tid + s2];
      rm[tid] = fmaxf(rm[tid], rm[tid + s2]);
    }
    __syncthreads();
  }
  if (tid == 0) {
    sums[bc] = rs[0];
    maxs[bc] = rm[0];
  }
}

struct FinalParams {
  const float* se1[2][3];
  const float* se2[2][3];
  const float* fc1w[2];
  const float* fc1b[2];
  const float* fc2w[2];
  const float* fc2b[2];
  const float* lw1;
  const float* lb1;
  const float* lw2;
  const float* lb2;
  const float* sums[2][3];
  const float* maxs[2][3];
};

__global__ void final_kernel(FinalParams P, float* __restrict__ out) {
  int b = threadIdx.x;
  if (b >= 8) return;
  float z16[16];
  for (int br = 0; br < 2; ++br) {
    int Cc = br ? 16 : 64;
    int rr = br ? 1 : 4;
    float zc[192];
    for (int ki = 0; ki < 3; ++ki) {
      int ks = 3 + ki;
      float invL = 1.f / (float)(3000 - ks + 1);
      const float* su = P.sums[br][ki] + b * Cc;
      const float* mx = P.maxs[br][ki] + b * Cc;
      float tt[4];
      for (int j = 0; j < rr; ++j) {
        float a = 0.f;
        for (int c = 0; c < Cc; ++c) a += (su[c] * invL) * P.se1[br][ki][j * Cc + c];
        tt[j] = a > 0.f ? a : 0.f;
      }
      for (int c = 0; c < Cc; ++c) {
        float a = 0.f;
        for (int j = 0; j < rr; ++j) a += tt[j] * P.se2[br][ki][c * rr + j];
        float y = 1.f / (1.f + __expf(-a));
        zc[ki * Cc + c] = mx[c] * y;
      }
    }
    float f1[64];
    for (int o = 0; o < Cc; ++o) {
      float a = P.fc1b[br][o];
      for (int k = 0; k < 3 * Cc; ++k) a += zc[k] * P.fc1w[br][o * 3 * Cc + k];
      f1[o] = leaky(a);
    }
    for (int o = 0; o < 8; ++o) {
      float a = P.fc2b[br][o];
      for (int k = 0; k < Cc; ++k) a += f1[k] * P.fc2w[br][o * Cc + k];
      z16[br * 8 + o] = a;
    }
  }
  float t8[8];
  for (int o = 0; o < 8; ++o) {
    float a = P.lb1[o];
    for (int k = 0; k < 16; ++k) a += z16[k] * P.lw1[o * 16 + k];
    t8[o] = leaky(a);
  }
  for (int o = 0; o < 2; ++o) {
    float a = P.lb2[o];
    for (int k = 0; k < 8; ++k) a += t8[k] * P.lw2[o * 8 + k];
    out[b * 2 + o] = a;
  }
}

// ---------------------------------------------------------------------------
// Host side
// ---------------------------------------------------------------------------
static void slopes_pow2(int m, float* out) {
  double start = pow(2.0, -pow(2.0, -(log2((double)m) - 3.0)));
  double v = start;
  for (int i = 0; i < m; ++i) { out[i] = (float)v; v *= start; }
}
static void get_slopes(int n, float* out) {
  if ((n & (n - 1)) == 0) { slopes_pow2(n, out); return; }
  int c = 1;
  while ((c << 1) <= n) c <<= 1;
  slopes_pow2(c, out);
  float big[32];
  get_slopes(2 * c, big);
  for (int i = 0; i < n - c; ++i) out[c + i] = big[2 * i];
}

static size_t attn_lds_bytes(int D) {
  int KP = (D < 32) ? 32 : D;
  int ST = KP + 8, SC = 168, CTS = 168, SKT = KP + 8;
  size_t f16cnt = (size_t)160 * ST + (size_t)160 * SC + (size_t)D * CTS * 2 + (size_t)D * SKT;
  size_t f32cnt = (size_t)D * D + 2 * 160 + 160 + 160;
  return f16cnt * 2 + f32cnt * 4;
}

extern "C" void kernel_launch(void* const* d_in, const int* in_sizes, int n_in,
                              void* d_out, int out_size, void* d_ws, size_t ws_size,
                              hipStream_t stream) {
  (void)in_sizes; (void)n_in; (void)out_size; (void)ws_size;
  auto F = [&](int i) { return (const float*)d_in[i]; };
  const float* x    = F(0);
  const float* seqx = F(1);
  const float* lt_g = F(2);  const float* lt_b = F(3);
  const float* lt_w1 = F(4); const float* lt_b1 = F(5);
  const float* lt_w2 = F(6); const float* lt_b2 = F(7);
  const float* ls_g = F(8);  const float* ls_b = F(9);
  const float* ls_w1 = F(10); const float* ls_b1 = F(11);
  const float* ls_w2 = F(12); const float* ls_b2 = F(13);
  const float* p1w = F(14); const float* p1b = F(15);
  const float* p2w = F(16); const float* p2b = F(17);
  const float* pe0 = F(18);
  const int CNN64 = 19, CNN16 = 35;

  float* ws = (float*)d_ws;
  const size_t BIG = 15360000;            // 8*10*3000*64
  float* R  = ws;                         // main residual
  float* X  = ws + BIG;                   // main state
  float* A  = ws + 2 * BIG;               // attn/ff scratch
  float* XM = ws + 3 * BIG;               // [8,3000,64]
  float* S0 = XM + 1536000;               // seq residual [8,4,3000,16]
  float* S1 = S0 + 1536000;               // seq state
  float* S2 = S1 + 1536000;               // seq scratch
  float* SM = S2 + 1536000;               // [8,3000,16]
  float* CS = SM + 384000;                // conv reductions
  float* sums64 = CS;                     // 3*512
  float* maxs64 = CS + 1536;              // 3*512
  float* sums16 = CS + 3072;              // 3*128
  float* maxs16 = CS + 3456;              // 3*128

  SlopeArr sl10 = {}, sl4 = {};
  get_slopes(10, sl10.s);
  get_slopes(4, sl4.s);

  const size_t ldsAttn64 = attn_lds_bytes(64);
  const size_t ldsAttn16 = attn_lds_bytes(16);
  const size_t ldsFFmain = (size_t)(384 * 72 + 64 * 392 + 64 * 72 + 64 * 392) * 2 + (384 + 64) * 4;
  const size_t ldsFFseq  = (size_t)(64 * 40 + 16 * 72 + 64 * 40 + 64 * 72) * 2 + (64 + 16) * 4;

  // ---------------- main branch: H=10, D=64 ----------------
  ln64_kernel<<<30000, 256, 0, stream>>>(x, nullptr, lt_g, lt_b, R, 240000);
  for (int it = 0; it < 2; ++it) {
    const float* q = (it == 0) ? R : X;
    attn_kernel<64><<<80, 256, ldsAttn64, stream>>>(q, A, sl10, 10);
    ln64_kernel<<<30000, 256, 0, stream>>>(A, R, lt_g, lt_b, X, 240000);
    ff_kernel<64, 64, 384, 64><<<3750, 256, ldsFFmain, stream>>>(X, lt_w1, lt_b1, lt_w2, lt_b2, A, 240000L);
    ln64_kernel<<<30000, 256, 0, stream>>>(A, R, lt_g, lt_b, X, 240000);
  }
  headmean_kernel<<<6000, 256, 0, stream>>>(X, XM, 10, 64, 1536000L);

  // ---------------- seq branch: H=4, D=16, seq_residual ----------------
  preproj_kernel<<<24000, 64, 0, stream>>>(seqx, p1w, p1b, p2w, p2b, pe0, S0);
  ln16_kernel<<<6000, 256, 0, stream>>>(S0, nullptr, ls_g, ls_b, S0, 96000);
  for (int it = 0; it < 2; ++it) {
    const float* q = (it == 0) ? S0 : S1;
    attn_kernel<16><<<32, 256, ldsAttn16, stream>>>(q, S2, sl4, 4);
    ln16_kernel<<<6000, 256, 0, stream>>>(S2, S0, ls_g, ls_b, S1, 96000);
    ff_kernel<16, 32, 64, 16><<<1500, 256, ldsFFseq, stream>>>(S1, ls_w1, ls_b1, ls_w2, ls_b2, S2, 96000L);
    ln16_kernel<<<6000, 256, 0, stream>>>(S2, S1, ls_g, ls_b, S1, 96000);
  }
  headmean_kernel<<<1500, 256, 0, stream>>>(S1, SM, 4, 16, 384000L);

  // ---------------- TextCNN branches ----------------
  for (int ki = 0; ki < 3; ++ki) {
    conv_reduce_kernel<<<8 * 64, 256, 0, stream>>>(XM, F(CNN64 + ki * 4 + 0), F(CNN64 + ki * 4 + 1),
                                                   sums64 + ki * 512, maxs64 + ki * 512, 64, 3000, 3 + ki);
    conv_reduce_kernel<<<8 * 16, 256, 0, stream>>>(SM, F(CNN16 + ki * 4 + 0), F(CNN16 + ki * 4 + 1),
                                                   sums16 + ki * 128, maxs16 + ki * 128, 16, 3000, 3 + ki);
  }

  FinalParams P;
  for (int ki = 0; ki < 3; ++ki) {
    P.se1[0][ki] = F(CNN64 + ki * 4 + 2);
    P.se2[0][ki] = F(CNN64 + ki * 4 + 3);
    P.se1[1][ki] = F(CNN16 + ki * 4 + 2);
    P.se2[1][ki] = F(CNN16 + ki * 4 + 3);
    P.sums[0][ki] = sums64 + ki * 512;
    P.maxs[0][ki] = maxs64 + ki * 512;
    P.sums[1][ki] = sums16 + ki * 128;
    P.maxs[1][ki] = maxs16 + ki * 128;
  }
  P.fc1w[0] = F(CNN64 + 12); P.fc1b[0] = F(CNN64 + 13);
  P.fc2w[0] = F(CNN64 + 14); P.fc2b[0] = F(CNN64 + 15);
  P.fc1w[1] = F(CNN16 + 12); P.fc1b[1] = F(CNN16 + 13);
  P.fc2w[1] = F(CNN16 + 14); P.fc2b[1] = F(CNN16 + 15);
  P.lw1 = F(51); P.lb1 = F(52); P.lw2 = F(53); P.lb2 = F(54);
  final_kernel<<<1, 32, 0, stream>>>(P, (float*)d_out);
}
